// DualEncoderLSTMDense_53077205844743
// MI455X (gfx1250) — matrix operand order repaired and checked by reading, not compiled, not run on hardware
//
#include <hip/hip_runtime.h>
#include <hip/hip_bf16.h>

// ---------------------------------------------------------------------------
// DualEncoderLSTMDense on gfx1250 (MI455X).
// Both encoders batched into one 512-row recurrence. Per timestep:
//   Z[512,2048] = concat(emb_gather, h)[512,768] x Wk[768,2048] + bias
// via v_wmma_f32_16x16x32_bf16 (bf16 in, f32 accum). Weights pre-converted
// and pre-transposed to [N][K] bf16; per-chunk weight tiles are staged to LDS
// by the Tensor Data Mover (tensor_load_to_lds + s_wait_tensorcnt).
// Then f32 peephole cell update, q_t = h_q x M (same WMMA kernel), tiny MLP.
// ---------------------------------------------------------------------------

typedef unsigned short u16;
typedef __bf16 v16bf __attribute__((ext_vector_type(16)));
typedef float  v8f   __attribute__((ext_vector_type(8)));
typedef unsigned int uint32x4 __attribute__((ext_vector_type(4)));
typedef int          int32x4  __attribute__((ext_vector_type(4)));
typedef int          int32x8  __attribute__((ext_vector_type(8)));

union BFrag { v16bf v; u16 u[16]; };

#define BATCH   256
#define SEQL    128
#define EMB     256
#define HID     512
#define FOURH   2048
#define KLSTM   768   // EMB + HID
#define MROWS   512   // 2 encoders * BATCH

// f32 -> bf16 round-to-nearest-even (bit twiddle)
__device__ inline u16 f2bf(float x) {
    union { float f; unsigned u; } p; p.f = x;
    unsigned u = p.u;
    unsigned r = (u + 0x7FFFu + ((u >> 16) & 1u)) >> 16;
    return (u16)r;
}

__device__ inline float sigmoidf(float x) { return 1.0f / (1.0f + __expf(-x)); }

// ---------------------------------------------------------------------------
// TDM load of one B tile: 128 rows x 32 bf16 elements, row stride K elements,
// into LDS at lds_off with rows packed contiguously (== Blds[n][k] layout).
// D# built per CDNA5 ISA ch.8 (group0/group1 bit layout); groups 2/3 zero
// (<= 2D tensor). Issued by one wave; completion via TENSORcnt.
// ---------------------------------------------------------------------------
__device__ inline void tdm_load_b_tile(const u16* gsrc, unsigned lds_off,
                                       int K, int N) {
    unsigned long long ga = (unsigned long long)(uintptr_t)gsrc;
    uint32x4 g0;
    g0.x = 1u;                                        // count=1 (valid), user mode
    g0.y = lds_off;                                   // lds_addr (bytes)
    g0.z = (unsigned)(ga & 0xFFFFFFFFu);              // global_addr[31:0]
    g0.w = (unsigned)((ga >> 32) & 0x01FFFFFFu)       // global_addr[56:32]
         | (2u << 30);                                // type=2 ("image")
    int32x8 g1;
    g1[0] = (int)(1u << 16);                          // data_size=1 (2 bytes)
    g1[1] = (int)((unsigned)K << 16);                 // tensor_dim0[15:0]
    g1[2] = (int)(((unsigned)K >> 16) |               // tensor_dim0[31:16]
                  ((unsigned)N << 16));               // tensor_dim1[15:0]
    g1[3] = (int)(((unsigned)N >> 16) |               // tensor_dim1[31:16]
                  (32u << 16));                       // tile_dim0 = 32
    g1[4] = 128;                                      // tile_dim1=128, tile_dim2=0
    g1[5] = K;                                        // tensor_dim0_stride[31:0]
    g1[6] = 0;                                        // stride0[47:32], dim1_stride lo
    g1[7] = 0;
    int32x4 z4 = {0, 0, 0, 0};
#if defined(__clang_major__) && (__clang_major__ >= 23)
    int32x8 z8 = {0, 0, 0, 0, 0, 0, 0, 0};
    __builtin_amdgcn_tensor_load_to_lds(g0, g1, z4, z4, z8, 0);
#else
    __builtin_amdgcn_tensor_load_to_lds(g0, g1, z4, z4, 0);
#endif
}

// ---------------------------------------------------------------------------
// WMMA GEMM: C[Mrows,N] = A[Mrows,K](f32->bf16) x BwT[N,K](bf16, TRANSPOSED)
//            (+ bias).
// AMODE==0: A is a plain f32 array.
// AMODE==1: LSTM step A: row R -> encoder e=R>>8, batch b=R&255;
//           k<256: embeddings[ids_e[b][t]][k], k>=256: hbuf[R][k-256].
//           The emb/h split is uniform per 32-wide K chunk (256 % 32 == 0).
// Tile: 128(M) x 128(N) per WG, K chunked by 32. 8 waves; each wave owns a
// 16x128 strip -> 8 f32 accumulators (64 VGPRs). B tiles staged by TDM.
// ---------------------------------------------------------------------------
template <int AMODE>
__global__ __launch_bounds__(256)
void gemm_wmma_bf16(const float* __restrict__ A, const u16* __restrict__ BwT,
                    const float* __restrict__ bias, float* __restrict__ C,
                    int N, int K,
                    const float* __restrict__ emb, const int* __restrict__ idsq,
                    const int* __restrict__ idsr, const float* __restrict__ hbuf,
                    int t) {
    const int tid = threadIdx.x;
    const int wgN = blockIdx.x * 128;
    const int wgM = blockIdx.y * 128;

    __shared__ __align__(16) u16 Alds[128 * 32];   // [row][k]  8 KB
    __shared__ __align__(16) u16 Blds[128 * 32];   // [n][k]    8 KB

    v8f acc[8];
#pragma unroll
    for (int i = 0; i < 8; ++i) acc[i] = (v8f){0.f,0.f,0.f,0.f,0.f,0.f,0.f,0.f};

    const int lane = tid & 31;
    const int wave = tid >> 5;
    // staging coordinates: each thread owns 16 consecutive k of one row
    const int srow = tid >> 1;
    const int sk0  = (tid & 1) * 16;
    const unsigned blds_off = (unsigned)(uintptr_t)(&Blds[0]);  // LDS byte offset

    for (int kc = 0; kc < K; kc += 32) {
        // ---- B tile [128 x 32] via Tensor Data Mover (wave 0 only) ---------
        if (wave == 0) {
            tdm_load_b_tile(BwT + (long)wgN * K + kc, blds_off, K, N);
        }
        // ---- stage A chunk [128 x 32]: 4x float4 load, cvt, 2x uint4 store -
        {
            const float* src;
            if (AMODE == 1) {
                int R = wgM + srow;
                int e = R >> 8, b = R & 255;
                if (kc < EMB) {
                    const int* ids = e ? idsr : idsq;
                    int tok = ids[b * SEQL + t];
                    src = emb + (long)tok * EMB + kc;
                } else {
                    src = hbuf + (long)R * HID + (kc - EMB);
                }
            } else {
                src = A + (long)(wgM + srow) * K + kc;
            }
            const float4* s4 = (const float4*)(src + sk0);
            float4 f0 = s4[0], f1 = s4[1], f2 = s4[2], f3 = s4[3];
            union { uint4 q; u16 u[8]; } p0, p1;
            p0.u[0] = f2bf(f0.x); p0.u[1] = f2bf(f0.y);
            p0.u[2] = f2bf(f0.z); p0.u[3] = f2bf(f0.w);
            p0.u[4] = f2bf(f1.x); p0.u[5] = f2bf(f1.y);
            p0.u[6] = f2bf(f1.z); p0.u[7] = f2bf(f1.w);
            p1.u[0] = f2bf(f2.x); p1.u[1] = f2bf(f2.y);
            p1.u[2] = f2bf(f2.z); p1.u[3] = f2bf(f2.w);
            p1.u[4] = f2bf(f3.x); p1.u[5] = f2bf(f3.y);
            p1.u[6] = f2bf(f3.z); p1.u[7] = f2bf(f3.w);
            uint4* d = (uint4*)&Alds[srow * 32 + sk0];
            d[0] = p0.q; d[1] = p1.q;
        }
        // wave 0 drains the TDM before releasing the barrier
        if (wave == 0) {
            __builtin_amdgcn_s_wait_tensorcnt(0);
        }
        __syncthreads();

        // ---- A fragment (16x32 bf16, ISA 7.12.2 layout) ----
        BFrag af;
        {
            int m   = lane & 15;
            int row = wave * 16 + m;
            int kb  = (lane >> 4) * 8;          // lanes 16-31 hold K+8
#pragma unroll
            for (int v2 = 0; v2 < 8; ++v2) {
                int kh = (v2 < 4) ? (kb + 2 * v2) : (16 + kb + 2 * (v2 - 4));
                af.u[2 * v2]     = Alds[row * 32 + kh];
                af.u[2 * v2 + 1] = Alds[row * 32 + kh + 1];
            }
        }
        // ---- 8 column tiles: B fragment + WMMA ----
#pragma unroll
        for (int nt = 0; nt < 8; ++nt) {
            BFrag bf;
            int n  = nt * 16 + (lane & 15);
            int ko = (lane >> 4) * 8;           // lanes 16-31 hold K+8
#pragma unroll
            for (int j = 0; j < 16; ++j) bf.u[j] = Blds[n * 32 + ((j < 8) ? (ko + j) : (16 + ko + j - 8))];
            acc[nt] = __builtin_amdgcn_wmma_f32_16x16x32_bf16(
                false, af.v, false, bf.v, (short)0, acc[nt], false, false);
        }
        __syncthreads();
    }

    // ---- epilogue: C/D lane layout (VGPR i -> M=i / M=8+i) + bias ----
#pragma unroll
    for (int nt = 0; nt < 8; ++nt) {
        int col = wgN + nt * 16 + (lane & 15);
        float bv = bias ? bias[col] : 0.0f;
#pragma unroll
        for (int i = 0; i < 8; ++i) {
            int row = wgM + wave * 16 + (lane >> 4) * 8 + i;
            C[(long)row * N + col] = acc[nt][i] + bv;
        }
    }
}

// ---------------------------------------------------------------------------
// Peephole LSTM cell update (gate order i,j,f,o; forget_bias = 2.0).
// State frozen when t >= sequence length (dynamic_rnn semantics).
// ---------------------------------------------------------------------------
__global__ __launch_bounds__(256)
void lstm_cell(const float* __restrict__ Z, float* __restrict__ c,
               float* __restrict__ h,
               const float* __restrict__ wi, const float* __restrict__ wf,
               const float* __restrict__ wo,
               const int* __restrict__ lenq, const int* __restrict__ lenr,
               int t) {
    int idx = blockIdx.x * 256 + threadIdx.x;        // 0 .. 512*512-1
    int R = idx >> 9;                                 // row (enc*256 + b)
    int u = idx & 511;                                // hidden unit
    int e = R >> 8, b = R & 255;
    int len = e ? lenr[b] : lenq[b];

    const float* z = Z + (long)R * FOURH;
    float zi = z[u], zj = z[HID + u], zf = z[2 * HID + u], zo = z[3 * HID + u];
    float cc = c[idx];
    float ig = sigmoidf(zi + wi[u] * cc);
    float fg = sigmoidf(zf + wf[u] * cc + 2.0f);
    float cn = fg * cc + ig * tanhf(zj);
    float og = sigmoidf(zo + wo[u] * cn);
    float hn = og * tanhf(cn);
    if (t < len) { c[idx] = cn; h[idx] = hn; }
}

// ---------------------------------------------------------------------------
// Final MLP head: 1280 rows of X = [q_t[qr] || r_h[rr]] -> relu(XW1+b1) ->
// relu(.W2+b2). One thread per output row (10 hidden units, K=1024).
// ---------------------------------------------------------------------------
__global__ __launch_bounds__(256)
void final_mlp(const float* __restrict__ qt, const float* __restrict__ hbuf,
               const float* __restrict__ W1, const float* __restrict__ b1,
               const float* __restrict__ W2, const float* __restrict__ b2,
               float* __restrict__ out) {
    int r = blockIdx.x * 256 + threadIdx.x;
    if (r >= 5 * BATCH) return;
    int qrow, rrow;
    if (r < BATCH) { qrow = r; rrow = r; }
    else {
        int m = r - BATCH;
        int o = (m >> 8) + 1;           // offs in 1..4, o-major layout
        int b = m & 255;
        qrow = b; rrow = (b + o) & 255; // (b+o) % 256
    }
    const float* q  = qt + (long)qrow * HID;
    const float* rh = hbuf + (long)(BATCH + rrow) * HID;  // replies = rows 256..511

    float h1[10];
#pragma unroll
    for (int j = 0; j < 10; ++j) h1[j] = b1[j];
    for (int k = 0; k < HID; ++k) {
        float x = q[k];
#pragma unroll
        for (int j = 0; j < 10; ++j) h1[j] += x * W1[k * 10 + j];
    }
    for (int k = 0; k < HID; ++k) {
        float x = rh[k];
#pragma unroll
        for (int j = 0; j < 10; ++j) h1[j] += x * W1[(HID + k) * 10 + j];
    }
    float lg = b2[0];
#pragma unroll
    for (int j = 0; j < 10; ++j) {
        float hv = h1[j] > 0.0f ? h1[j] : 0.0f;
        lg += hv * W2[j];
    }
    out[r] = lg > 0.0f ? lg : 0.0f;
}

// ---------------------------------------------------------------------------
// Prep kernels
// ---------------------------------------------------------------------------
// d[n*K + k] = bf16(s[k*N + n])  (transpose + convert; one-time, L2 resident)
__global__ void transpose_f32_bf16(const float* __restrict__ s, u16* __restrict__ d,
                                   int K, int N) {
    int total = K * N;
    for (int i = blockIdx.x * blockDim.x + threadIdx.x; i < total;
         i += gridDim.x * blockDim.x) {
        int n = i / K, k = i - n * K;
        d[i] = f2bf(s[(long)k * N + n]);
    }
}
__global__ void zero_buf(float* __restrict__ p, int n) {
    for (int i = blockIdx.x * blockDim.x + threadIdx.x; i < n; i += gridDim.x * blockDim.x)
        p[i] = 0.0f;
}

// ---------------------------------------------------------------------------
// Workspace layout (bytes, all 256B aligned): total 10 MB
// ---------------------------------------------------------------------------
static const size_t O_WK = 0;                          // WkT [2048][768] bf16 = 3145728
static const size_t O_M  = 3145728;                    // MT  [512][512]  bf16 =  524288
static const size_t O_H  = 3670016;                    // h [512][512] f32 = 1048576
static const size_t O_C  = 4718592;                    // c [512][512] f32 = 1048576
static const size_t O_Z  = 5767168;                    // Z [512][2048] f32 = 4194304
static const size_t O_QT = 9961472;                    // qt [256][512] f32 = 524288

extern "C" void kernel_launch(void* const* d_in, const int* in_sizes, int n_in,
                              void* d_out, int out_size, void* d_ws, size_t ws_size,
                              hipStream_t stream) {
    const float* emb   = (const float*)d_in[0];
    const float* Wk    = (const float*)d_in[1];
    const float* bias  = (const float*)d_in[2];
    const float* wi    = (const float*)d_in[3];
    const float* wf    = (const float*)d_in[4];
    const float* wo    = (const float*)d_in[5];
    const float* Mmat  = (const float*)d_in[6];
    const float* W1    = (const float*)d_in[7];
    const float* b1    = (const float*)d_in[8];
    const float* W2    = (const float*)d_in[9];
    const float* b2    = (const float*)d_in[10];
    const int*   idsq  = (const int*)d_in[11];
    const int*   idsr  = (const int*)d_in[12];
    const int*   lenq  = (const int*)d_in[13];
    const int*   lenr  = (const int*)d_in[14];

    char* ws = (char*)d_ws;
    u16*   WkT  = (u16*)(ws + O_WK);   // [FOURH][KLSTM]
    u16*   MT   = (u16*)(ws + O_M);    // [HID][HID]
    float* h    = (float*)(ws + O_H);
    float* c    = (float*)(ws + O_C);
    float* Z    = (float*)(ws + O_Z);
    float* qt   = (float*)(ws + O_QT);
    float* out  = (float*)d_out;

    // Prep: transposed bf16 weight copies + zero state (h,c contiguous)
    transpose_f32_bf16<<<512, 256, 0, stream>>>(Wk, WkT, KLSTM, FOURH);
    transpose_f32_bf16<<<256, 256, 0, stream>>>(Mmat, MT, HID, HID);
    zero_buf<<<512, 256, 0, stream>>>(h, 2 * MROWS * HID);

    // Recurrence: both encoders fused (512 rows)
    for (int t = 0; t < SEQL; ++t) {
        gemm_wmma_bf16<1><<<dim3(16, 4), 256, 0, stream>>>(
            nullptr, WkT, bias, Z, FOURH, KLSTM, emb, idsq, idsr, h, t);
        lstm_cell<<<1024, 256, 0, stream>>>(Z, c, h, wi, wf, wo, lenq, lenr, t);
    }

    // q_t = h_q (rows 0..255) x M   [256,512]x[512,512]
    gemm_wmma_bf16<0><<<dim3(4, 2), 256, 0, stream>>>(
        h, MT, nullptr, qt, HID, HID, nullptr, nullptr, nullptr, nullptr, 0);

    // MLP head -> logits [1280,1]
    final_mlp<<<5, 256, 0, stream>>>(qt, h, W1, b1, W2, b2, out);
}
